// CrashingVids_17987323036155
// MI455X (gfx1250) — matrix operand order repaired and checked by reading, not compiled, not run on hardware
//
#include <hip/hip_runtime.h>
#include <hip/hip_bf16.h>

// ---------- types ----------
typedef __attribute__((ext_vector_type(16))) __bf16 v16bf;
typedef __attribute__((ext_vector_type(8)))  __bf16 v8bf;
typedef __attribute__((ext_vector_type(8)))  float  v8f;
typedef __attribute__((ext_vector_type(4)))  unsigned int u32x4;
typedef __attribute__((ext_vector_type(8)))  int i32x8;
typedef __attribute__((ext_vector_type(4)))  int i32x4;

union AFrag { v16bf v; v8bf h[2]; };

#define HAVE_TDM __has_builtin(__builtin_amdgcn_tensor_load_to_lds)

// ---------- problem constants ----------
#define BB   16
#define TT   750
#define TP   752           // padded time (zero row front+back)
#define DD   2048
#define CC   20
#define MM   12000         // B*T
#define KTOT 6144          // 3*2048
#define KEASY 150
#define KHARD 37

// output offsets (floats, reference tuple order)
#define OFF_VID   0LL
#define OFF_EA    320LL
#define OFF_EB    4915520LL
#define OFF_HA    9830720LL
#define OFF_HB    11043136LL
#define OFF_ACT   12255552LL
#define OFF_CAS   12267552LL

#if HAVE_TDM
__device__ inline unsigned cv_lds_off(void* p) {
    return (unsigned)(unsigned long long)(__attribute__((address_space(3))) char*)p;
}
#endif

// ---------- prep: x -> padded bf16 [B][752][2048] ----------
__global__ __launch_bounds__(256)
void cv_xpad_kernel(const float* __restrict__ x, __bf16* __restrict__ xpad) {
    long long i = (long long)blockIdx.x * 256 + threadIdx.x;
    if (i >= (long long)BB * TP * DD) return;
    int ii = (int)(i & (DD - 1));
    long long row = i >> 11;                 // b*752 + tt
    int tt = (int)(row % TP);
    long long b = row / TP;
    float v = 0.0f;
    if (tt >= 1 && tt <= TT)
        v = x[(b * TT + (tt - 1)) * DD + ii];
    xpad[i] = (__bf16)v;
}

// ---------- prep: w1 [O][I][3] -> bf16 B-matrix wt[o][k*2048+i] ----------
__global__ __launch_bounds__(256)
void cv_wt_kernel(const float* __restrict__ w1, __bf16* __restrict__ wt) {
    long long i = (long long)blockIdx.x * 256 + threadIdx.x;
    if (i >= (long long)DD * 3 * DD) return;
    int ii = (int)(i & (DD - 1));
    long long r = i >> 11;                   // o*3 + k
    int k = (int)(r % 3);
    long long o = r / 3;
    wt[i] = (__bf16)w1[(o * DD + ii) * 3 + k];
}

// ---------- main conv-as-GEMM with bf16 WMMA ----------
// block tile 256x128, 8 waves of 32, wave tile 64(M) x 64(N), K chunk 32
// double-buffered LDS; B tile staged by the Tensor Data Mover when available
__global__ __launch_bounds__(256)
void cv_gemm_kernel(const __bf16* __restrict__ xpad, const __bf16* __restrict__ wt,
                    const float* __restrict__ b1, float* __restrict__ h) {
    __shared__ __align__(64) __bf16 lA[2][256][32];   // 2 x 16KB
    __shared__ __align__(64) __bf16 lB[2][128][32];   // 2 x 8KB

    const int tid  = threadIdx.x;
    const int lane = tid & 31;
    const int wid  = tid >> 5;
    const int wm   = wid & 3;            // 4 waves along M (64 rows each)
    const int wn   = wid >> 2;           // 2 waves along N (64 cols each)
    const int rowBase = blockIdx.x * 256;
    const int colBase = blockIdx.y * 128;

    // A tile cooperative load coords: 1 thread per row, 32 bf16 (64B) each
    const int ar = tid;
    const int am = rowBase + ar;
    const bool avalid = (am < MM);
    int ab = am / TT, at = am - ab * TT;
    const long long abase = avalid ? ((long long)(ab * TP + at)) * DD : 0;

    // B tile fallback cooperative load coords
    const int br = tid >> 1;
    const int bh = (tid & 1) * 16;
    const long long bbase = ((long long)(colBase + br)) * KTOT;

    v8f acc[4][4] = {};

    // fragment read coords (A and B both contiguous k = 16*(lane/16) + s)
    const int arow = wm * 64 + (lane & 15);
    const int kb8  = (lane >> 4) * 8;
    const int bcol = wn * 64 + (lane & 15);
    const int kb16 = (lane >> 4) * 16;

    auto stageA = [&](int kk0, int buf) {
        const int tap = kk0 >> 11;
        const int i0  = kk0 & (DD - 1);
        v16bf a0, a1;
        if (avalid) {
            const __bf16* p = xpad + abase + (long long)tap * DD + i0;
            a0 = *(const v16bf*)(p);
            a1 = *(const v16bf*)(p + 16);
        } else {
            #pragma unroll
            for (int q = 0; q < 16; q++) { a0[q] = (__bf16)0.0f; a1[q] = (__bf16)0.0f; }
        }
        *(v16bf*)&lA[buf][ar][0]  = a0;
        *(v16bf*)&lA[buf][ar][16] = a1;
    };

    auto stageB = [&](int kk0, int buf) {
#if HAVE_TDM
        if (wid == 0) {
            // Tensor DMA: 2D tile, 128 rows x 32 bf16, row stride 6144 elems
            unsigned long long ga =
                (unsigned long long)(const void*)(wt + (long long)colBase * KTOT + kk0);
            u32x4 g0;
            g0.x = 1u;                                   // count=1 (valid descriptor)
            g0.y = cv_lds_off(&lB[buf][0][0]);           // lds_addr (bytes)
            g0.z = (unsigned)(ga & 0xffffffffull);       // global_addr lo
            g0.w = (unsigned)((ga >> 32) & 0x01ffffffull) | 0x80000000u; // hi | type=2
            i32x8 g1;
            g1[0] = 1 << 16;                             // data_size = 1 (2 bytes)
            g1[1] = (int)(6144u << 16);                  // tensor_dim0 = 6144 (lo16)
            g1[2] = (int)((6144u >> 16) | (128u << 16)); // dim0 hi | tensor_dim1 = 128
            g1[3] = (int)(32u << 16);                    // tensor_dim1 hi | tile_dim0 = 32
            g1[4] = 128;                                 // tile_dim1 = 128, tile_dim2 = 0
            g1[5] = 6144;                                // tensor_dim0_stride lo
            g1[6] = 0;                                   // stride hi | dim1_stride lo
            g1[7] = 0;
            __builtin_amdgcn_tensor_load_to_lds(g0, g1, (i32x4)0, (i32x4)0, (i32x8)0, 0);
        }
#else
        *(v16bf*)&lB[buf][br][bh] = *(const v16bf*)(wt + bbase + kk0 + bh);
#endif
    };

    // prologue: stage chunk 0
    stageA(0, 0);
    stageB(0, 0);
#if HAVE_TDM
    if (wid == 0) __builtin_amdgcn_s_wait_tensorcnt(0);
#endif
    __syncthreads();

    const int NIT = KTOT / 32;       // 192
    for (int it = 0; it < NIT; ++it) {
        const int cur = it & 1, nxt = cur ^ 1;
        if (it + 1 < NIT) {
            stageA((it + 1) * 32, nxt);
            stageB((it + 1) * 32, nxt);
        }

        AFrag af[4];
        #pragma unroll
        for (int tm = 0; tm < 4; tm++) {
            const int r = arow + tm * 16;
            af[tm].h[0] = *(const v8bf*)&lA[cur][r][kb16];
            af[tm].h[1] = *(const v8bf*)&lA[cur][r][kb16 + 8];
        }
        v16bf bf[4];
        #pragma unroll
        for (int tn = 0; tn < 4; tn++)
            bf[tn] = *(const v16bf*)&lB[cur][bcol + tn * 16][kb16];

        #pragma unroll
        for (int tm = 0; tm < 4; tm++)
            #pragma unroll
            for (int tn = 0; tn < 4; tn++)
                acc[tm][tn] = __builtin_amdgcn_wmma_f32_16x16x32_bf16(
                    false, af[tm].v, false, bf[tn], (short)0, acc[tm][tn],
                    false, false);

#if HAVE_TDM
        if (wid == 0) __builtin_amdgcn_s_wait_tensorcnt(0);
#endif
        __syncthreads();
    }

    // epilogue: bias + relu; C layout: N = lane&15, elem v -> M = v + 8*(lane>=16)
    const int nlo = lane & 15;
    const int mhi = (lane >> 4) * 8;
    #pragma unroll
    for (int tm = 0; tm < 4; tm++)
        #pragma unroll
        for (int tn = 0; tn < 4; tn++) {
            const int col = colBase + wn * 64 + tn * 16 + nlo;
            const float bias = b1[col];
            #pragma unroll
            for (int v = 0; v < 8; v++) {
                const int m = rowBase + wm * 64 + tm * 16 + v + mhi;
                if (m < MM) {
                    float val = acc[tm][tn][v] + bias;
                    h[(long long)m * DD + col] = val > 0.0f ? val : 0.0f;
                }
            }
        }
}

// ---------- cas = relu(h @ w2^T), actionness = sum_c cas ----------
__global__ __launch_bounds__(128)
void cv_cas_kernel(const float* __restrict__ h, const float* __restrict__ w2,
                   float* __restrict__ cas, float* __restrict__ act) {
    __shared__ float part[128 * CC + CC];
    const int m = blockIdx.x;
    const int tid = threadIdx.x;
    const float* hr = h + (long long)m * DD;
    float a[CC];
    #pragma unroll
    for (int c = 0; c < CC; c++) a[c] = 0.0f;
    for (int o = tid; o < DD; o += 128) {
        const float hv = hr[o];
        #pragma unroll
        for (int c = 0; c < CC; c++) a[c] += hv * w2[c * DD + o];
    }
    #pragma unroll
    for (int c = 0; c < CC; c++) part[tid * CC + c] = a[c];
    __syncthreads();
    if (tid < CC) {
        float s = 0.0f;
        for (int i = 0; i < 128; i++) s += part[i * CC + tid];
        s = s > 0.0f ? s : 0.0f;
        cas[(long long)m * CC + tid] = s;
        part[128 * CC + tid] = s;
    }
    __syncthreads();
    if (tid == 0) {
        float s = 0.0f;
        #pragma unroll
        for (int c = 0; c < CC; c++) s += part[128 * CC + c];
        act[m] = s;
    }
}

// ---------- per-batch: max, median(stable ranks), erode/dilate, 4 score arrays ----------
__global__ __launch_bounds__(256)
void cv_analysis_kernel(const float* __restrict__ act, float* __restrict__ scores) {
    __shared__ float a[TT];
    __shared__ float bin[TT];
    __shared__ float red[256];
    __shared__ float medsh;
    const int b = blockIdx.x, tid = threadIdx.x;
    for (int t = tid; t < TT; t += 256) a[t] = act[b * TT + t];
    if (tid == 0) medsh = 0.0f;
    __syncthreads();
    float mx = -3.4e38f;
    for (int t = tid; t < TT; t += 256) mx = fmaxf(mx, a[t]);
    red[tid] = mx;
    __syncthreads();
    for (int s = 128; s > 0; s >>= 1) {
        if (tid < s) red[tid] = fmaxf(red[tid], red[tid + s]);
        __syncthreads();
    }
    const float maxv = red[0];
    // median = mean of ascending order stats 374,375 (stable rank)
    for (int t = tid; t < TT; t += 256) {
        const float v = a[t];
        int r = 0;
        for (int j = 0; j < TT; j++) {
            const float u = a[j];
            r += (u < v) || (u == v && j < t);
        }
        if (r == 374 || r == 375) atomicAdd(&medsh, v * 0.5f);
    }
    __syncthreads();
    const float med = medsh;
    for (int t = tid; t < TT; t += 256) bin[t] = a[t] > med ? 1.0f : 0.0f;
    __syncthreads();
    for (int t = tid; t < TT; t += 256) {
        auto gb = [&](int i) -> float { return (i >= 0 && i < TT) ? bin[i] : 0.0f; };
        float e3 = fminf(gb(t - 1), fminf(gb(t), gb(t + 1)));
        float e6 = e3;
        e6 = fminf(e6, gb(t - 3)); e6 = fminf(e6, gb(t - 2)); e6 = fminf(e6, gb(t + 2));
        float d3 = fmaxf(gb(t - 1), fmaxf(gb(t), gb(t + 1)));
        float d6 = d3;
        d6 = fmaxf(d6, gb(t - 2)); d6 = fmaxf(d6, gb(t + 2)); d6 = fmaxf(d6, gb(t + 3));
        const float inner = e3 - e6, outer = d6 - d3;
        const float av = a[t];
        scores[(0 * BB + b) * TT + t] = av;          // easy_act
        scores[(1 * BB + b) * TT + t] = maxv - av;   // easy_bkg
        scores[(2 * BB + b) * TT + t] = av * inner;  // hard_act
        scores[(3 * BB + b) * TT + t] = av * outer;  // hard_bkg
    }
}

// ---------- stable descending top-k + gather embedding rows ----------
__global__ __launch_bounds__(256)
void cv_topk_gather_kernel(const float* __restrict__ scores, const float* __restrict__ h,
                           float* __restrict__ out) {
    __shared__ float s[TT];
    __shared__ int idx[KEASY];
    const int b = blockIdx.x, sel = blockIdx.y, tid = threadIdx.x;
    const int k = (sel < 2) ? KEASY : KHARD;
    for (int t = tid; t < TT; t += 256) s[t] = scores[(sel * BB + b) * TT + t];
    __syncthreads();
    for (int t = tid; t < TT; t += 256) {
        const float v = s[t];
        int r = 0;
        for (int j = 0; j < TT; j++) {
            const float u = s[j];
            r += (u > v) || (u == v && j < t);   // stable argsort(-s)
        }
        if (r < k) idx[r] = t;
    }
    __syncthreads();
    long long base = (sel == 0) ? OFF_EA : (sel == 1) ? OFF_EB : (sel == 2) ? OFF_HA : OFF_HB;
    for (int p = 0; p < k; p++) {
        const int t = idx[p];
        const float4* src = (const float4*)(h + (long long)(b * TT + t) * DD);
        float4* dst = (float4*)(out + base + (long long)(b * k + p) * DD);
        for (int q = tid; q < DD / 4; q += 256) dst[q] = src[q];
    }
}

// ---------- video scores: per (b,c) top-150 column sum ----------
__global__ __launch_bounds__(256)
void cv_vtopk_kernel(const float* __restrict__ cas, float* __restrict__ means) {
    __shared__ float col[TT];
    __shared__ float red[256];
    const int b = blockIdx.x, c = blockIdx.y, tid = threadIdx.x;
    for (int t = tid; t < TT; t += 256) col[t] = cas[(long long)(b * TT + t) * CC + c];
    __syncthreads();
    float part = 0.0f;
    for (int t = tid; t < TT; t += 256) {
        const float v = col[t];
        int r = 0;
        for (int j = 0; j < TT; j++) {
            const float u = col[j];
            r += (u > v) || (u == v && j < t);
        }
        if (r < KEASY) part += v;
    }
    red[tid] = part;
    __syncthreads();
    for (int s = 128; s > 0; s >>= 1) {
        if (tid < s) red[tid] += red[tid + s];
        __syncthreads();
    }
    if (tid == 0) means[b * CC + c] = red[0] / (float)KEASY;
}

// ---------- softmax over classes ----------
__global__ __launch_bounds__(32)
void cv_softmax_kernel(const float* __restrict__ means, float* __restrict__ out) {
    const int b = blockIdx.x;
    if (threadIdx.x == 0) {
        float v[CC], mx = -3.4e38f;
        #pragma unroll
        for (int c = 0; c < CC; c++) { v[c] = means[b * CC + c]; mx = fmaxf(mx, v[c]); }
        float s = 0.0f;
        #pragma unroll
        for (int c = 0; c < CC; c++) { v[c] = expf(v[c] - mx); s += v[c]; }
        #pragma unroll
        for (int c = 0; c < CC; c++) out[OFF_VID + b * CC + c] = v[c] / s;
    }
}

extern "C" void kernel_launch(void* const* d_in, const int* in_sizes, int n_in,
                              void* d_out, int out_size, void* d_ws, size_t ws_size,
                              hipStream_t stream) {
    const float* x  = (const float*)d_in[0];
    const float* w1 = (const float*)d_in[1];
    const float* b1 = (const float*)d_in[2];
    const float* w2 = (const float*)d_in[3];
    float* out = (float*)d_out;
    char* ws = (char*)d_ws;

    __bf16* xpad  = (__bf16*)(ws);                   // 16*752*2048*2   = 49,283,072 B
    __bf16* wt    = (__bf16*)(ws + 49283072);        // 2048*6144*2     = 25,165,824 B
    float*  h     = (float*)(ws + 74448896);         // 12000*2048*4    = 98,304,000 B
    float*  sc    = (float*)(ws + 172752896);        // 4*16*750*4      = 192,000 B
    float*  means = (float*)(ws + 172944896);        // 320*4

    float* cas_out = out + OFF_CAS;
    float* act_out = out + OFF_ACT;

    cv_xpad_kernel<<<(BB * TP * DD) / 256, 256, 0, stream>>>(x, xpad);
    cv_wt_kernel<<<(DD * 3 * DD) / 256, 256, 0, stream>>>(w1, wt);

    dim3 gg((MM + 255) / 256, DD / 128);             // 47 x 16
    cv_gemm_kernel<<<gg, 256, 0, stream>>>(xpad, wt, b1, h);

    cv_cas_kernel<<<MM, 128, 0, stream>>>(h, w2, cas_out, act_out);
    cv_analysis_kernel<<<BB, 256, 0, stream>>>(act_out, sc);
    dim3 tg(BB, 4);
    cv_topk_gather_kernel<<<tg, 256, 0, stream>>>(sc, h, out);
    dim3 vg(BB, CC);
    cv_vtopk_kernel<<<vg, 256, 0, stream>>>(cas_out, means);
    cv_softmax_kernel<<<BB, 32, 0, stream>>>(means, out);
}
